// OptiXAttention_7464653161189
// MI455X (gfx1250) — matrix operand order repaired and checked by reading, not compiled, not run on hardware
//
#include <hip/hip_runtime.h>
#include <cstddef>

// ---------------------------------------------------------------------------
// Causal MHA forward for MI455X (gfx1250, wave32, WMMA).
// B=2, S=2048, E=1024, H=16, D=64.  All matmuls on v_wmma_f32_16x16x32_f16.
// GEMM weight tiles staged through LDS with async-to-LDS (ASYNCcnt) copies.
// ---------------------------------------------------------------------------

typedef __attribute__((ext_vector_type(16))) _Float16 v16h;
typedef __attribute__((ext_vector_type(8)))  _Float16 v8h;
typedef __attribute__((ext_vector_type(8)))  float    v8f;
typedef __attribute__((ext_vector_type(4)))  int      v4i;

union hvec16 { v16h v; v8h h8[2]; };

#define WMMA_F16(A, B, C) \
  __builtin_amdgcn_wmma_f32_16x16x32_f16(false, (A), false, (B), (short)0, (C), false, false)

static constexpr int   kS     = 2048;
static constexpr int   kE     = 1024;
static constexpr int   kH     = 16;
static constexpr int   kD     = 64;
static constexpr int   kM     = 2 * kS;    // 4096 rows (B*S)
static constexpr float kScale = 0.125f;    // 1/sqrt(64)

// ----------------------------- conversion kernels --------------------------

__global__ __launch_bounds__(256) void cvt_f32_to_f16(const float* __restrict__ src,
                                                      _Float16* __restrict__ dst, int n) {
  int i = blockIdx.x * blockDim.x + threadIdx.x;
  int stride = gridDim.x * blockDim.x;
  for (; i < n; i += stride) dst[i] = (_Float16)src[i];
}

// W is (K=1024, N=1024) row-major fp32; write Wt (N-major) f16: Wt[n*1024+k].
__global__ __launch_bounds__(256) void cvt_transpose_f16(const float* __restrict__ W,
                                                         _Float16* __restrict__ Wt) {
  int i = blockIdx.x * blockDim.x + threadIdx.x;
  if (i >= kE * kE) return;
  int k = i >> 10, n = i & 1023;
  Wt[(size_t)n * kE + k] = (_Float16)W[i];
}

// ----------------------------- generic WMMA GEMM ---------------------------
// C(M=4096, N=1024) = A(M,1024)f16 @ Wt(N,1024)f16^T + bias.
// 256 threads = 8 waves; wave w owns a 32x64 strip, block covers 256x64.
// The shared 64x32 B tile per K-step is staged into LDS once per workgroup
// using GLOBAL_LOAD_ASYNC_TO_LDS_B128 (double-buffered, ASYNCcnt-tracked).
// MODE 0: Q  -> f16 (B,H,S,D), scaled by kScale
// MODE 1: K  -> f16 (B,H,S,D)
// MODE 2: V  -> f16 (B,H,D,S)   (transposed for the PV WMMA B-fragments)
// MODE 3: fp32 row-major (final output projection)

template <int MODE>
__global__ __launch_bounds__(256) void gemm_wmma(const _Float16* __restrict__ A,
                                                 const _Float16* __restrict__ Wt,
                                                 const float* __restrict__ bias,
                                                 void* __restrict__ out) {
  __shared__ __align__(16) _Float16 btile[2][64 * 32];   // 2 x 4KB double buffer

  const int tid  = threadIdx.x;
  const int wave = tid >> 5;
  const int lane = tid & 31;
  const int lo   = lane & 15;
  const int hi   = lane >> 4;

  const int m0 = blockIdx.y * 256 + wave * 32;
  const int n0 = blockIdx.x * 64;

  // staging map: 256 threads x 16B cover the 64x32-half tile
  const int snl = tid >> 2;         // local col 0..63
  const int sj  = (tid & 3) * 8;    // k-chunk (halves)
  const _Float16* sg_base = Wt + (size_t)(n0 + snl) * kE + sj;
  _Float16* sl[2] = { &btile[0][snl * 32 + sj], &btile[1][snl * 32 + sj] };

  auto stage = [&](int kb, int buf) {
    const _Float16* g = sg_base + kb;
    _Float16* l = sl[buf];
#if __has_builtin(__builtin_amdgcn_global_load_async_to_lds_b128)
    typedef __attribute__((address_space(1))) v4i* as1_v4i;
    typedef __attribute__((address_space(3))) v4i* as3_v4i;
    __builtin_amdgcn_global_load_async_to_lds_b128(
        (as1_v4i)(v4i*)g, (as3_v4i)(v4i*)l, 0, 0);
#else
    *(v8h*)l = *(const v8h*)g;
#endif
  };

  v8f acc[2][4] = {};

  const _Float16* ar0 = A + (size_t)(m0 + lo) * kE;
  const _Float16* ar1 = A + (size_t)(m0 + 16 + lo) * kE;

  stage(0, 0);
  for (int kb = 0; kb < kE; kb += 32) {
    const int buf = (kb >> 5) & 1;
    if (kb + 32 < kE) {
      stage(kb + 32, buf ^ 1);
      asm volatile("s_wait_asynccnt 0x1" ::: "memory");   // tile `buf` landed
    } else {
      asm volatile("s_wait_asynccnt 0x0" ::: "memory");
    }
    __syncthreads();   // every thread's slice of tile `buf` is in LDS

    hvec16 af0, af1;
    af0.h8[0] = *(const v8h*)(ar0 + kb + hi * 8);        // K = hi*8 .. +7
    af0.h8[1] = *(const v8h*)(ar0 + kb + 16 + hi * 8);   // K = 16+hi*8 .. +7
    af1.h8[0] = *(const v8h*)(ar1 + kb + hi * 8);
    af1.h8[1] = *(const v8h*)(ar1 + kb + 16 + hi * 8);

    const _Float16* bb = &btile[buf][0];
#pragma unroll
    for (int t = 0; t < 4; ++t) {
      hvec16 bf;
      const _Float16* lp = bb + (t * 16 + lo) * 32 + hi * 8;
      bf.h8[0] = *(const v8h*)lp;
      bf.h8[1] = *(const v8h*)(lp + 16);
      acc[0][t] = WMMA_F16(af0.v, bf.v, acc[0][t]);
      acc[1][t] = WMMA_F16(af1.v, bf.v, acc[1][t]);
    }
    __syncthreads();   // all reads of `buf` done before it is re-staged
  }

#pragma unroll
  for (int u = 0; u < 2; ++u) {
#pragma unroll
    for (int t = 0; t < 4; ++t) {
      const int n  = n0 + t * 16 + lo;
      const float bn = bias[n];
#pragma unroll
      for (int r = 0; r < 8; ++r) {
        const int m = m0 + u * 16 + r + 8 * hi;
        float v = acc[u][t][r] + bn;
        if (MODE == 0 || MODE == 1) {
          if (MODE == 0) v *= kScale;
          const int b = m >> 11, s = m & 2047, h = n >> 6, d = n & 63;
          ((_Float16*)out)[((size_t)(b * kH + h) * kS + s) * kD + d] = (_Float16)v;
        } else if (MODE == 2) {
          const int b = m >> 11, s = m & 2047, h = n >> 6, d = n & 63;
          ((_Float16*)out)[((size_t)(b * kH + h) * kD + d) * kS + s] = (_Float16)v;
        } else {
          ((float*)out)[(size_t)m * kE + n] = v;
        }
      }
    }
  }
}

// ----------------------------- flash attention -----------------------------

__device__ __forceinline__ float rowmax16(float v) {
  v = fmaxf(v, __shfl_xor(v, 1, 16));
  v = fmaxf(v, __shfl_xor(v, 2, 16));
  v = fmaxf(v, __shfl_xor(v, 4, 16));
  v = fmaxf(v, __shfl_xor(v, 8, 16));
  return v;
}
__device__ __forceinline__ float rowsum16(float v) {
  v += __shfl_xor(v, 1, 16);
  v += __shfl_xor(v, 2, 16);
  v += __shfl_xor(v, 4, 16);
  v += __shfl_xor(v, 8, 16);
  return v;
}

// One wave per 32-query tile (two 16-row WMMA tiles): each K/V fragment load
// feeds two score / two PV WMMAs.  8 independent waves per workgroup.
// Q,K: (B,H,S,D) f16 (Q pre-scaled by 1/sqrt(D)); Vt: (B,H,D,S) f16.
// ctx: (B,S,E) f16 row-major, E index = h*64 + d.
__global__ __launch_bounds__(256) void attn_kernel(const _Float16* __restrict__ Q,
                                                   const _Float16* __restrict__ K,
                                                   const _Float16* __restrict__ Vt,
                                                   _Float16* __restrict__ ctx) {
  __shared__ __align__(16) _Float16 ptile[8][2][16][32];   // per-wave P staging (16KB)

  const int wave = threadIdx.x >> 5;
  const int lane = threadIdx.x & 31;
  const int lo   = lane & 15;
  const int hi   = lane >> 4;

  const int tile = blockIdx.x * 8 + wave;   // 0..2047
  const int bh   = tile >> 6;               // (b*H + h); 64 query-groups per head
  const int q0   = (tile & 63) << 5;        // 32 queries per wave
  const int b    = bh >> 4;
  const int h    = bh & 15;

  const _Float16* __restrict__ Qp = Q  + (size_t)bh * kS * kD;
  const _Float16* __restrict__ Kp = K  + (size_t)bh * kS * kD;
  const _Float16* __restrict__ Vp = Vt + (size_t)bh * kD * kS;

  // Q A-fragments: [qt][0] covers D 0..31, [qt][1] covers D 32..63.
  hvec16 qf[2][2];
#pragma unroll
  for (int qt = 0; qt < 2; ++qt) {
    const _Float16* rp = Qp + (size_t)(q0 + qt * 16 + lo) * kD;
    qf[qt][0].h8[0] = *(const v8h*)(rp +      hi * 16);
    qf[qt][0].h8[1] = *(const v8h*)(rp + 8  + hi * 16);
    qf[qt][1].h8[0] = *(const v8h*)(rp + 32 + hi * 16);
    qf[qt][1].h8[1] = *(const v8h*)(rp + 40 + hi * 16);
  }

  v8f O[2][4] = {};
  float Mrow[2][8], Lrow[2][8];
#pragma unroll
  for (int qt = 0; qt < 2; ++qt)
#pragma unroll
    for (int r = 0; r < 8; ++r) { Mrow[qt][r] = -__builtin_inff(); Lrow[qt][r] = 0.f; }

  const int kblocks = (q0 >> 5) + 1;   // causal: blocks up to and incl. the diagonal
  for (int jb = 0; jb < kblocks; ++jb) {
    const int key0 = jb << 5;

    // K B-fragments for this 32-key block (shared by both query tiles)
    hvec16 kb0[2], kb1[2];
#pragma unroll
    for (int sub = 0; sub < 2; ++sub) {
      const _Float16* kp = Kp + (size_t)(key0 + sub * 16 + lo) * kD + hi * 16;
      kb0[sub].v = *(const v16h*)kp;          // D 0..31
      kb1[sub].v = *(const v16h*)(kp + 32);   // D 32..63
    }

    float alpha[2][8];
#pragma unroll
    for (int qt = 0; qt < 2; ++qt) {
      // scores: S(16x32) = Q(16x64) @ K^T, two 16x16 WMMA chains over D
      v8f s[2];
#pragma unroll
      for (int sub = 0; sub < 2; ++sub) {
        v8f z = {};
        v8f t0 = WMMA_F16(qf[qt][0].v, kb0[sub].v, z);
        s[sub] = WMMA_F16(qf[qt][1].v, kb1[sub].v, t0);
      }

      // causal mask (Q already carries the 1/sqrt(D) scale)
#pragma unroll
      for (int sub = 0; sub < 2; ++sub) {
        const int keyn = key0 + sub * 16 + lo;
#pragma unroll
        for (int r = 0; r < 8; ++r) {
          const int q = q0 + qt * 16 + r + 8 * hi;
          if (keyn > q) s[sub][r] = -__builtin_inff();
        }
      }

      // online softmax per row slot (rows striped across 16 lanes per half-wave)
#pragma unroll
      for (int r = 0; r < 8; ++r) {
        const float mx = fmaxf(rowmax16(s[0][r]), rowmax16(s[1][r]));
        const float Mn = fmaxf(Mrow[qt][r], mx);
        alpha[qt][r] = __expf(Mrow[qt][r] - Mn);
        Mrow[qt][r]  = Mn;
        const float p0 = __expf(s[0][r] - Mn);
        const float p1 = __expf(s[1][r] - Mn);
        s[0][r] = p0;
        s[1][r] = p1;
        Lrow[qt][r] = Lrow[qt][r] * alpha[qt][r] + rowsum16(p0) + rowsum16(p1);
      }

      // C-layout -> A-layout relayout of P through LDS (f16 16x32 tile)
#pragma unroll
      for (int sub = 0; sub < 2; ++sub)
#pragma unroll
        for (int r = 0; r < 8; ++r)
          ptile[wave][qt][r + 8 * hi][sub * 16 + lo] = (_Float16)s[sub][r];
    }

    asm volatile("s_wait_dscnt 0" ::: "memory");   // DS ops are in-order per wave
    hvec16 pf[2];
#pragma unroll
    for (int qt = 0; qt < 2; ++qt) {
      pf[qt].h8[0] = *(const v8h*)&ptile[wave][qt][lo][hi * 16];
      pf[qt].h8[1] = *(const v8h*)&ptile[wave][qt][lo][8 + hi * 16];
    }
    asm volatile("s_wait_dscnt 0" ::: "memory");

    // rescale running output, then O += P @ V (V pre-transposed: (B,H,D,S))
#pragma unroll
    for (int qt = 0; qt < 2; ++qt)
#pragma unroll
      for (int t = 0; t < 4; ++t)
#pragma unroll
        for (int r = 0; r < 8; ++r) O[qt][t][r] *= alpha[qt][r];

#pragma unroll
    for (int t = 0; t < 4; ++t) {
      hvec16 bv;
      bv.v = *(const v16h*)(Vp + (size_t)(t * 16 + lo) * kS + key0 + hi * 16);
      O[0][t] = WMMA_F16(pf[0].v, bv.v, O[0][t]);
      O[1][t] = WMMA_F16(pf[1].v, bv.v, O[1][t]);
    }
  }

  // normalize and store context (f16, (B,S,E) row-major)
#pragma unroll
  for (int qt = 0; qt < 2; ++qt) {
    float rinv[8];
#pragma unroll
    for (int r = 0; r < 8; ++r) rinv[r] = (Lrow[qt][r] > 0.f) ? (1.f / Lrow[qt][r]) : 0.f;
#pragma unroll
    for (int t = 0; t < 4; ++t) {
#pragma unroll
      for (int r = 0; r < 8; ++r) {
        const int q   = q0 + qt * 16 + r + 8 * hi;
        const int col = h * kD + t * 16 + lo;
        ctx[((size_t)(b * kS + q)) * kE + col] = (_Float16)(O[qt][t][r] * rinv[r]);
      }
    }
  }
}

// ----------------------------- launcher ------------------------------------

extern "C" void kernel_launch(void* const* d_in, const int* in_sizes, int n_in,
                              void* d_out, int out_size, void* d_ws, size_t ws_size,
                              hipStream_t stream) {
  (void)in_sizes; (void)n_in; (void)out_size; (void)ws_size;

  const float* x  = (const float*)d_in[0];
  const float* Wq = (const float*)d_in[1];
  const float* bq = (const float*)d_in[2];
  const float* Wk = (const float*)d_in[3];
  const float* bk = (const float*)d_in[4];
  const float* Wv = (const float*)d_in[5];
  const float* bv = (const float*)d_in[6];
  const float* Wo = (const float*)d_in[7];
  const float* bo = (const float*)d_in[8];

  // workspace carve-up (halves): 48 MB total
  _Float16* ws    = (_Float16*)d_ws;
  const size_t MTOK = (size_t)kM * kE;       // 4M elements
  const size_t WTOK = (size_t)kE * kE;       // 1M elements
  _Float16* x16   = ws;
  _Float16* wq16t = x16 + MTOK;
  _Float16* wk16t = wq16t + WTOK;
  _Float16* wv16t = wk16t + WTOK;
  _Float16* wo16t = wv16t + WTOK;
  _Float16* q16   = wo16t + WTOK;
  _Float16* k16   = q16 + MTOK;
  _Float16* v16t  = k16 + MTOK;
  _Float16* ctx16 = v16t + MTOK;

  cvt_f32_to_f16<<<2048, 256, 0, stream>>>(x, x16, (int)MTOK);
  cvt_transpose_f16<<<4096, 256, 0, stream>>>(Wq, wq16t);
  cvt_transpose_f16<<<4096, 256, 0, stream>>>(Wk, wk16t);
  cvt_transpose_f16<<<4096, 256, 0, stream>>>(Wv, wv16t);
  cvt_transpose_f16<<<4096, 256, 0, stream>>>(Wo, wo16t);

  dim3 gg(16, 16);   // N/64 x M/256
  gemm_wmma<0><<<gg, 256, 0, stream>>>(x16, wq16t, bq, q16);
  gemm_wmma<1><<<gg, 256, 0, stream>>>(x16, wk16t, bk, k16);
  gemm_wmma<2><<<gg, 256, 0, stream>>>(x16, wv16t, bv, v16t);

  attn_kernel<<<256, 256, 0, stream>>>(q16, k16, v16t, ctx16);

  gemm_wmma<3><<<gg, 256, 0, stream>>>(ctx16, wo16t, bo, d_out);
}
